// Attention_8564164788953
// MI455X (gfx1250) — compile-verified, matrix operand order checked by reading, not run on hardware
//
#include <hip/hip_runtime.h>
#include <hip/hip_bf16.h>

#define WSZ   16
#define NTOK  256
#define DIM   128
#define NH    4
#define HD    32
#define NWAVE 8

typedef __attribute__((ext_vector_type(16))) _Float16 v16h;
typedef __attribute__((ext_vector_type(8)))  _Float16 v8h;
typedef __attribute__((ext_vector_type(8)))  float    v8f;

// LDS row strides (in halfs); all multiples of 8 so fragment chunks are 16B aligned
#define XS_LD 136
#define W_LD  136
#define QK_LD 40
#define VT_LD 264
#define P_LD  264
#define O_LD  136

#define WMMA(a, b, c) \
  __builtin_amdgcn_wmma_f32_16x16x32_f16(false, (a), false, (b), (short)0, (c), false, false)

// A-style fragment (also B when the K-major operand is stored transposed):
// lane (hl = lane>>4, ln = lane&15) holds row (r0+ln), K halves
// {k0+8*hl .. +7} and {k0+16+8*hl .. +7}  -> two 16-byte LDS loads.
__device__ __forceinline__ v16h lds_frag(const _Float16* base, int ld, int r0, int k0, int lane) {
  const int hl = lane >> 4;
  const _Float16* p = base + (r0 + (lane & 15)) * ld + k0 + hl * 8;
  v8h lo = *(const v8h*)(p);
  v8h hi = *(const v8h*)(p + 16);
  v16h o;
#pragma unroll
  for (int j = 0; j < 8; ++j) { o[j] = lo[j]; o[j + 8] = hi[j]; }
  return o;
}

// B fragment gathered straight from a K-major f32 global weight matrix.
__device__ __forceinline__ v16h gmem_frag_kmajor(const float* W, int ldw, int k0, int col, int lane) {
  const int hl = lane >> 4;
  v16h o;
#pragma unroll
  for (int j = 0; j < 8; ++j) {
    o[j]     = (_Float16)W[(k0 + hl * 8 + j) * ldw + col];
    o[j + 8] = (_Float16)W[(k0 + 16 + hl * 8 + j) * ldw + col];
  }
  return o;
}

__global__ __launch_bounds__(256, 1)
void win_attn_fused(const float* __restrict__ x, const float* __restrict__ noise,
                    const float* __restrict__ qkv_w, const float* __restrict__ proj_w,
                    const float* __restrict__ proj_b, const float* __restrict__ bias_table,
                    const float* __restrict__ noise_strength, float* __restrict__ out) {
  __shared__ __align__(16) _Float16 Xs  [NTOK * XS_LD];       // x (f16), one window
  __shared__ __align__(16) _Float16 WTh [96 * W_LD];          // per-head qkv weight slice, col-major
  __shared__ __align__(16) _Float16 Qh  [NTOK * QK_LD];
  __shared__ __align__(16) _Float16 Kh  [NTOK * QK_LD];
  __shared__ __align__(16) _Float16 VhT [HD * VT_LD];         // V transposed: [hd][tok]
  __shared__ __align__(16) _Float16 Pbuf[NWAVE * 16 * P_LD];  // per-wave softmaxed P strip
  __shared__ __align__(16) _Float16 Obuf[NTOK * O_LD];        // concat-head attention output
  __shared__ float biasS[(2 * WSZ - 1) * (2 * WSZ - 1) * NH];

  const int tid  = threadIdx.x;
  const int lane = tid & 31;
  const int wv   = tid >> 5;
  const int hl   = lane >> 4;
  const int ln   = lane & 15;
  const int bwin = blockIdx.x;

  // ---------------- Phase 0: stage bias table + noised input as f16 ----------------
  for (int e = tid; e < (2 * WSZ - 1) * (2 * WSZ - 1) * NH; e += 256) biasS[e] = bias_table[e];

  const float ns = noise_strength[0];
  const float4* x4 = (const float4*)(x + (size_t)bwin * NTOK * DIM);
#pragma unroll
  for (int it = 0; it < 32; ++it) {
    int e4  = tid + it * 256;        // 8192 float4s per window
    int tok = e4 >> 5;               // 32 float4 per row
    int c4  = (e4 & 31) << 2;
    float4 v = x4[e4];
    float nz = noise[bwin * NTOK + tok] * ns;
    _Float16* d = &Xs[tok * XS_LD + c4];
    d[0] = (_Float16)(v.x + nz); d[1] = (_Float16)(v.y + nz);
    d[2] = (_Float16)(v.z + nz); d[3] = (_Float16)(v.w + nz);
  }
  __syncthreads();

  const float scale = 0.17677669529663689f;  // 1/sqrt(32)

  for (int h = 0; h < NH; ++h) {
    // ------------- stage this head's qkv weight slice, column-major f16 -------------
    // c in [0,96): matrix = c/32 (Q,K,V), global col = matrix*128 + h*32 + (c%32)
    for (int e = tid; e < 96 * DIM; e += 256) {
      int c = e % 96, k = e / 96;
      int gcol = (c >> 5) * DIM + h * HD + (c & 31);
      WTh[c * W_LD + k] = (_Float16)qkv_w[k * (3 * DIM) + gcol];
    }
    __syncthreads();

    // ------------- QKV GEMM: 16x6 tiles; wave does rows {2wv,2wv+1} x 6 cols -------------
#pragma unroll
    for (int ct = 0; ct < 6; ++ct) {
      v8f acc0 = {}; v8f acc1 = {};
#pragma unroll
      for (int ks = 0; ks < 4; ++ks) {
        v16h bf = lds_frag(WTh, W_LD, ct * 16, ks * 32, lane);   // B via transposed weights
        v16h a0 = lds_frag(Xs, XS_LD, (2 * wv) * 16, ks * 32, lane);
        v16h a1 = lds_frag(Xs, XS_LD, (2 * wv + 1) * 16, ks * 32, lane);
        acc0 = WMMA(a0, bf, acc0);
        acc1 = WMMA(a1, bf, acc1);
      }
      int mat = ct >> 1;
      int nl  = (ct & 1) * 16 + ln;  // 0..31 inside Q/K/V
#pragma unroll
      for (int mi = 0; mi < 2; ++mi) {
        v8f acc = mi ? acc1 : acc0;
        int mbase = (2 * wv + mi) * 16 + hl * 8;
#pragma unroll
        for (int r = 0; r < 8; ++r) {
          int m = mbase + r;
          _Float16 val = (_Float16)acc[r];
          if (mat == 0)      Qh[m * QK_LD + nl]  = val;
          else if (mat == 1) Kh[m * QK_LD + nl]  = val;
          else               VhT[nl * VT_LD + m] = val;   // store V transposed
        }
      }
    }
    __syncthreads();

    // ------------- attention: each wave owns two 16-row query strips -------------
    _Float16* Pw = &Pbuf[wv * 16 * P_LD];
#pragma unroll 1
    for (int si = 0; si < 2; ++si) {
      const int m0 = (wv + si * 8) * 16;
      const int mrow0 = m0 + hl * 8;
      v16h qa = lds_frag(Qh, QK_LD, m0, 0, lane);  // K dim = 32 -> single fragment

      v8f sf[16];  // full [16 x 256] logit strip in registers
#pragma unroll
      for (int nt = 0; nt < 16; ++nt) {
        v16h kb = lds_frag(Kh, QK_LD, nt * 16, 0, lane);  // B = K^T via K rows
        v8f z = {};
        sf[nt] = WMMA(qa, kb, z);
      }
      // scale + relative-position bias (closed-form index, gathered from LDS)
#pragma unroll
      for (int nt = 0; nt < 16; ++nt) {
        int n = nt * 16 + ln;
        int rj = n >> 4, cj = n & 15;
#pragma unroll
        for (int r = 0; r < 8; ++r) {
          int m = mrow0 + r;
          int ri = m >> 4, ci = m & 15;
          int idx = (ri - rj + WSZ - 1) * (2 * WSZ - 1) + (ci - cj + WSZ - 1);
          sf[nt][r] = sf[nt][r] * scale + biasS[idx * NH + h];
        }
      }
      // row max across 16 tiles then across the 16-lane half (C-frag row layout)
      float vmax[8];
#pragma unroll
      for (int r = 0; r < 8; ++r) {
        float mx = sf[0][r];
#pragma unroll
        for (int nt = 1; nt < 16; ++nt) mx = fmaxf(mx, sf[nt][r]);
        mx = fmaxf(mx, __shfl_xor(mx, 1, 32));
        mx = fmaxf(mx, __shfl_xor(mx, 2, 32));
        mx = fmaxf(mx, __shfl_xor(mx, 4, 32));
        mx = fmaxf(mx, __shfl_xor(mx, 8, 32));
        vmax[r] = mx;
      }
      // exp, row sum, write P (f16) to this wave's private strip
      float vsum[8];
#pragma unroll
      for (int r = 0; r < 8; ++r) vsum[r] = 0.f;
#pragma unroll
      for (int nt = 0; nt < 16; ++nt) {
#pragma unroll
        for (int r = 0; r < 8; ++r) {
          float p = __expf(sf[nt][r] - vmax[r]);
          vsum[r] += p;
          Pw[(hl * 8 + r) * P_LD + nt * 16 + ln] = (_Float16)p;
        }
      }
      float rl[8];
#pragma unroll
      for (int r = 0; r < 8; ++r) {
        float sm = vsum[r];
        sm += __shfl_xor(sm, 1, 32);
        sm += __shfl_xor(sm, 2, 32);
        sm += __shfl_xor(sm, 4, 32);
        sm += __shfl_xor(sm, 8, 32);
        rl[r] = 1.0f / sm;
      }
      // O strip = P @ V   (B via V^T rows -> vectorized fragments)
      v8f oc0 = {}; v8f oc1 = {};
#pragma unroll
      for (int kt = 0; kt < 8; ++kt) {
        v16h pa  = lds_frag(Pw, P_LD, 0, kt * 32, lane);
        v16h vb0 = lds_frag(VhT, VT_LD, 0, kt * 32, lane);
        v16h vb1 = lds_frag(VhT, VT_LD, 16, kt * 32, lane);
        oc0 = WMMA(pa, vb0, oc0);
        oc1 = WMMA(pa, vb1, oc1);
      }
#pragma unroll
      for (int r = 0; r < 8; ++r) {
        int m = mrow0 + r;
        Obuf[m * O_LD + h * HD + ln]      = (_Float16)(oc0[r] * rl[r]);
        Obuf[m * O_LD + h * HD + 16 + ln] = (_Float16)(oc1[r] * rl[r]);
      }
    }
    __syncthreads();  // protect Qh/Kh/VhT before next head rewrites them
  }

  // ---------------- final projection: wave wv owns output column tile wv ----------------
  {
    const int col = wv * 16 + ln;
    v16h bp[4];
#pragma unroll
    for (int ks = 0; ks < 4; ++ks) bp[ks] = gmem_frag_kmajor(proj_w, DIM, ks * 32, col, lane);
    const float pb = proj_b[col];
    float* outw = out + (size_t)bwin * NTOK * DIM;
#pragma unroll 1
    for (int mt = 0; mt < 16; ++mt) {
      v8f acc = {};
#pragma unroll
      for (int ks = 0; ks < 4; ++ks) {
        v16h a = lds_frag(Obuf, O_LD, mt * 16, ks * 32, lane);
        acc = WMMA(a, bp[ks], acc);
      }
#pragma unroll
      for (int r = 0; r < 8; ++r) {
        int m = mt * 16 + hl * 8 + r;
        outw[m * DIM + col] = acc[r] + pb;
      }
    }
  }
}

extern "C" void kernel_launch(void* const* d_in, const int* in_sizes, int n_in,
                              void* d_out, int out_size, void* d_ws, size_t ws_size,
                              hipStream_t stream) {
  const float* x   = (const float*)d_in[0];
  const float* nz  = (const float*)d_in[1];
  const float* qw  = (const float*)d_in[2];
  const float* pw  = (const float*)d_in[3];
  const float* pb  = (const float*)d_in[4];
  const float* bt  = (const float*)d_in[5];
  const float* nst = (const float*)d_in[6];
  float* o = (float*)d_out;
  hipLaunchKernelGGL(win_attn_fused, dim3(1024), dim3(256), 0, stream,
                     x, nz, qw, pw, pb, bt, nst, o);
}
